// LocationEncoder_12670153523572
// MI455X (gfx1250) — compile-verified, matrix operand order checked by reading, not run on hardware
//
#include <hip/hip_runtime.h>
#include <stdint.h>

typedef __attribute__((ext_vector_type(16))) _Float16 v16h;
typedef __attribute__((ext_vector_type(8)))  _Float16 v8h;
typedef __attribute__((ext_vector_type(8)))  float    v8f;

#define BB   4
#define CCTX 512
#define TT   512
#define HH   128

// WMMA 16-bit fragment: element i of lane `lane` holds K = i + (i&8) + (lane>=16 ? 8 : 0).
// => elements 0..7 and 8..15 are K-contiguous 16-byte runs starting at klo and 16+klo.
__device__ __forceinline__ v16h cat16(v8h lo, v8h hi) {
    return __builtin_shufflevector(lo, hi, 0,1,2,3,4,5,6,7,8,9,10,11,12,13,14,15);
}

// CDNA5 async global->LDS copy (GVS mode): VDST = LDS byte address VGPR,
// VADDR = unsigned byte offset VGPR, SADDR = 64-bit base SGPR pair. ASYNCcnt-tracked.
__device__ __forceinline__ void async_b128(uint32_t lds_addr, const void* base, uint32_t off) {
    asm volatile("global_load_async_to_lds_b128 %0, %1, %2"
                 :: "v"(lds_addr), "v"(off), "s"(base) : "memory");
}
__device__ __forceinline__ void wait_async0() {
    asm volatile("s_wait_asynccnt 0x0" ::: "memory");
}

// ---------------------------------------------------------------------------
// Prep: key/query (f32, row-major) and value stored TRANSPOSED as f16:
//   VhT[(b*HH + h)*CCTX + c] = value[b,c,h]   (K-innermost for WMMA B-frags)
// ---------------------------------------------------------------------------
__global__ __launch_bounds__(128) void prep_kernel(
    const float* __restrict__ cx, const float* __restrict__ cy,
    const float* __restrict__ tx,
    const float* __restrict__ W_in, const float* __restrict__ b_in,
    const float* __restrict__ W_ctx, const float* __restrict__ b_ctx,
    float* __restrict__ Kf, float* __restrict__ Qf, _Float16* __restrict__ VhT)
{
    const int row = blockIdx.x;          // b*512 + r
    const int b   = row >> 9;
    const int r   = row & 511;
    const int h   = threadIdx.x;         // 0..127

    const float x0 = cx[row*4+0], x1 = cx[row*4+1], x2 = cx[row*4+2];
    const float y0 = cy[row*2+0], y1 = cy[row*2+1];
    const float u0 = tx[row*4+0], u1 = tx[row*4+1], u2 = tx[row*4+2];

    float v = b_in[h] + x0*W_in[0*HH+h] + x1*W_in[1*HH+h] + x2*W_in[2*HH+h]
                      + y0*W_in[3*HH+h] + y1*W_in[4*HH+h];
    VhT[(b*HH + h)*CCTX + r] = (_Float16)v;

    Kf[row*HH + h] = b_ctx[h] + x0*W_ctx[0*HH+h] + x1*W_ctx[1*HH+h] + x2*W_ctx[2*HH+h];
    Qf[row*HH + h] = b_ctx[h] + u0*W_ctx[0*HH+h] + u1*W_ctx[1*HH+h] + u2*W_ctx[2*HH+h];
}

// W_tgt (128x128, [k][n]) -> f16 transposed WtT[n*HH + k]  (K-innermost)
__global__ __launch_bounds__(256) void cvt_kernel(
    const float* __restrict__ W, _Float16* __restrict__ WtT)
{
    int i = blockIdx.x*256 + threadIdx.x;   // linear over output
    if (i < HH*HH) {
        int n = i >> 7, k = i & 127;
        WtT[i] = (_Float16)W[k*HH + n];
    }
}

// ---------------------------------------------------------------------------
// Fused: L1 scores -> softmax -> (w @ value) WMMA -> (@ W_tgt + b) WMMA.
// One block per (b, 16-target tile); 256 threads = 8 wave32, one N-tile/wave.
// Tiles double-buffered and staged with global_load_async_to_lds_b128.
// ---------------------------------------------------------------------------
__global__ __launch_bounds__(256) void attn_kernel(
    const float* __restrict__ Kf, const float* __restrict__ Qf,
    const _Float16* __restrict__ VhT, const _Float16* __restrict__ WtT,
    const float* __restrict__ b_tgt, float* __restrict__ out)
{
    __shared__ __align__(16) float    qs[16*HH];         // 8 KB  query rows
    __shared__ __align__(16) float    ktile[2*32*HH];    // 32 KB: 2 key bufs / 2 f16 value bufs
    __shared__ __align__(16) _Float16 w16[16*CCTX];      // 16 KB exp(s-max)
    __shared__ __align__(16) _Float16 rep16[16*HH];      // 4 KB
    __shared__ float                  pred[16*16];
    __shared__ float                  rowmax[16];
    __shared__ float                  rowrcp[16];

    const int tid  = threadIdx.x;
    const int bidx = blockIdx.x;
    const int b    = bidx >> 5;
    const int t0   = (bidx & 31) << 4;

    const uint32_t ktile_lds = (uint32_t)(uintptr_t)ktile;   // LDS byte offset

    // stage the 16 query rows (plain vector loads)
    {
        const float4* qsrc = (const float4*)(Qf + (b*TT + t0)*HH);
        float4* qdst = (float4*)qs;
        #pragma unroll
        for (int j = 0; j < 2; ++j)
            qdst[tid + j*256] = qsrc[tid + j*256];
    }

    const int t  = tid >> 4;                 // target row 0..15
    const int cg = tid & 15;                 // context group 0..15

    // ---- scores: key tiles (32 rows, 16 KB) async-staged, double buffered
    const float* KfB = Kf + b*CCTX*HH;
    #pragma unroll
    for (int j = 0; j < 4; ++j)
        async_b128(ktile_lds + (tid + j*256)*16, KfB, (tid + j*256)*16);

    float sc[32];
    for (int kt = 0; kt < 16; ++kt) {
        wait_async0();
        __syncthreads();
        if (kt < 15) {
            const uint32_t dst = ktile_lds + (uint32_t)(((kt+1)&1)*16384);
            const uint32_t go  = (uint32_t)((kt+1)*16384);
            #pragma unroll
            for (int j = 0; j < 4; ++j)
                async_b128(dst + (tid + j*256)*16, KfB, go + (tid + j*256)*16);
        }
        const float* kb = ktile + (kt&1)*4096;
        #pragma unroll
        for (int e = 0; e < 2; ++e) {
            const float* kp = kb + (cg*2 + e)*HH;
            const float* qp = qs + t*HH;
            float s = 0.f;
            #pragma unroll 8
            for (int h = 0; h < HH; ++h) s += fabsf(kp[h] - qp[h]);
            sc[kt*2 + e] = -0.5f * s;
        }
    }

    // ---- softmax: row max, unnormalized exp -> w16 (f16), reciprocal row sums
    float m = -1e30f;
    #pragma unroll
    for (int i = 0; i < 32; ++i) m = fmaxf(m, sc[i]);
    pred[t*16 + cg] = m;
    __syncthreads();
    if (cg == 0) {
        float rm = pred[t*16];
        #pragma unroll
        for (int i = 1; i < 16; ++i) rm = fmaxf(rm, pred[t*16+i]);
        rowmax[t] = rm;
    }
    __syncthreads();
    {
        const float rm = rowmax[t];
        float ssum = 0.f;
        #pragma unroll
        for (int i = 0; i < 32; ++i) {
            float e = __expf(sc[i] - rm);
            ssum += e;
            w16[t*CCTX + (i >> 1)*32 + cg*2 + (i & 1)] = (_Float16)e;
        }
        pred[t*16 + cg] = ssum;
    }
    __syncthreads();
    if (cg == 0) {
        float rs = 0.f;
        #pragma unroll
        for (int i = 0; i < 16; ++i) rs += pred[t*16+i];
        rowrcp[t] = 1.0f / rs;               // one divide per row, multiply later
    }
    __syncthreads();

    // ---- GEMM1: rep(16x128) = w(16x512) @ value(512x128), f16 WMMA, f32 acc
    const int wave = tid >> 5;
    const int lane = tid & 31;
    const int col  = (wave << 4) + (lane & 15);   // N coordinate
    const int mrow = lane & 15;                   // A row
    const int klo  = (lane & 16) >> 1;            // 0 or 8

    // value tile (transposed [n][k], 32 K-slab, 8 KB) async-staged, double buffered
    const _Float16* VtB = VhT + b*HH*CCTX;
    #pragma unroll
    for (int j = 0; j < 2; ++j) {
        const int c = tid + j*256;                // chunk: n = c>>2, koff = (c&3)*8
        async_b128(ktile_lds + c*16, VtB, (uint32_t)(((c>>2)*CCTX + (c&3)*8)*2));
    }

    v8f acc = {0.f,0.f,0.f,0.f,0.f,0.f,0.f,0.f};
    for (int kt = 0; kt < 16; ++kt) {
        wait_async0();
        __syncthreads();
        if (kt < 15) {
            const uint32_t dst = ktile_lds + (uint32_t)(((kt+1)&1)*8192);
            #pragma unroll
            for (int j = 0; j < 2; ++j) {
                const int c = tid + j*256;
                async_b128(dst + c*16, VtB,
                           (uint32_t)(((c>>2)*CCTX + (kt+1)*32 + (c&3)*8)*2));
            }
        }
        const _Float16* vb = (const _Float16*)ktile + (kt&1)*4096;
        // A frag: two 16B LDS loads; B frag: two 16B LDS loads
        const v8h* ap = (const v8h*)(w16 + mrow*CCTX + kt*32 + klo);
        const v8h* bp = (const v8h*)(vb + col*32 + klo);
        v16h a  = cat16(ap[0], ap[2]);
        v16h bm = cat16(bp[0], bp[2]);
        acc = __builtin_amdgcn_wmma_f32_16x16x32_f16(false, a, false, bm,
                                                     (short)0, acc, false, false);
    }

    // normalize by softmax denominator (multiply by precomputed reciprocal)
    #pragma unroll
    for (int r = 0; r < 8; ++r) {
        const int mr = r + ((lane >> 4) << 3);
        rep16[mr*HH + col] = (_Float16)(acc[r] * rowrcp[mr]);
    }
    __syncthreads();

    // ---- GEMM2: out(16x128) = rep(16x128) @ W_tgt(128x128) + b_tgt
    const float bias = b_tgt[col];
    v8f acc2 = {bias,bias,bias,bias,bias,bias,bias,bias};
    #pragma unroll
    for (int kt = 0; kt < 4; ++kt) {
        const v8h* ap = (const v8h*)(rep16 + mrow*HH + kt*32 + klo);   // LDS, 2x16B
        const v8h* bp = (const v8h*)(WtT + col*HH + kt*32 + klo);      // global, 2x16B
        v16h a2 = cat16(ap[0], ap[2]);
        v16h b2 = cat16(bp[0], bp[2]);
        acc2 = __builtin_amdgcn_wmma_f32_16x16x32_f16(false, a2, false, b2,
                                                      (short)0, acc2, false, false);
    }
    #pragma unroll
    for (int r = 0; r < 8; ++r) {
        const int mr = r + ((lane >> 4) << 3);
        out[(b*TT + t0 + mr)*HH + col] = acc2[r];
    }
}

// ---------------------------------------------------------------------------
extern "C" void kernel_launch(void* const* d_in, const int* in_sizes, int n_in,
                              void* d_out, int out_size, void* d_ws, size_t ws_size,
                              hipStream_t stream) {
    const float* cx    = (const float*)d_in[0];
    const float* cy    = (const float*)d_in[1];
    const float* tx    = (const float*)d_in[2];
    const float* W_in  = (const float*)d_in[3];
    const float* b_in  = (const float*)d_in[4];
    const float* W_ctx = (const float*)d_in[5];
    const float* b_ctx = (const float*)d_in[6];
    const float* W_tgt = (const float*)d_in[7];
    const float* b_tgt = (const float*)d_in[8];
    float* out = (float*)d_out;

    char* ws = (char*)d_ws;
    float*    Kf  = (float*)(ws);                                // 1 MB
    float*    Qf  = (float*)(ws + (1u << 20));                   // 1 MB
    _Float16* VhT = (_Float16*)(ws + (2u << 20));                // 512 KB (transposed)
    _Float16* WtT = (_Float16*)(ws + (2u << 20) + (512u << 10)); // 32 KB (transposed)

    prep_kernel<<<BB*CCTX, 128, 0, stream>>>(cx, cy, tx, W_in, b_in,
                                             W_ctx, b_ctx, Kf, Qf, VhT);
    cvt_kernel<<<(HH*HH + 255)/256, 256, 0, stream>>>(W_tgt, WtT);
    attn_kernel<<<BB*(TT/16), 256, 0, stream>>>(Kf, Qf, VhT, WtT, b_tgt, out);
}
